// EGNNpool_60730837565922
// MI455X (gfx1250) — matrix operand order repaired and checked by reading, not compiled, not run on hardware
//
#include <hip/hip_runtime.h>
#include <hip/hip_bf16.h>

// ---------------------------------------------------------------------------
// EGNN + pooling for MI455X (gfx1250, wave32, WMMA).
//
// Edge-MLP GEMMs dominate (~139 GFLOP); all operands are cache-resident, so
// we drive v_wmma_f32_16x16x32_f16. One wave == one 16-row tile. The
// first-GEMM weight pack (shared by all waves of a block) is staged into LDS
// once per block via gfx1250 async copies (global_load_async_to_lds_b128 +
// s_wait_asynccnt), so WMMA B operands come from ds_load_b128 instead of
// per-tile redundant global loads that were serializing on s_wait_loadcnt.
//
// Input-pointer order assumption (jax tree flatten: positional dict order for
// the top-level setup_inputs() dict, sorted keys inside `params`):
//   0:x 1:edge_index 2:edge_attr 3:batch
//   4:ein_b 5:ein_w 6:eout_b 7:eout_w
//   8+11*i .. : layer i = [Wc1, Wc2, We1, We2, Wn1, Wn2, bc1, be1, be2, bn1, bn2]
//   52:lin_b 53:lin_w
// ---------------------------------------------------------------------------

typedef _Float16 v16h __attribute__((ext_vector_type(16)));
typedef _Float16 v8h  __attribute__((ext_vector_type(8)));
typedef float    v8f  __attribute__((ext_vector_type(8)));
typedef int      v4i  __attribute__((ext_vector_type(4)));

constexpr int N_NODES  = 8192;
constexpr int N_EDGES  = 262144;
constexpr int N_GRAPHS = 64;
constexpr int HID      = 128;
constexpr int N_EF     = 4;
constexpr int N_PRED   = 8;
constexpr int N_LAYERS = 4;
constexpr int NT       = HID / 16;     // 8 N-tiles of 16 columns
constexpr int FRAG_H   = 32 * 16;      // halves per (kc,nt) fragment group
constexpr int WPB      = 4;            // waves per block (edge/node kernels)

struct Frag {
  union { v16h v; v8h h[2]; _Float16 e[16]; };
};

__device__ __forceinline__ float silu_f(float x) {
  return x / (1.0f + __expf(-x));
}

__device__ __forceinline__ v8h ld8h(const _Float16* p) {
  return *reinterpret_cast<const v8h*>(p);
}

// Pre-packed B fragment: one aligned 32B v16h per lane per (kc, nt).
__device__ __forceinline__ v16h ldB(const _Float16* Wp, int kc, int nt, int lane) {
  return *(reinterpret_cast<const v16h*>(Wp) + ((kc * NT + nt) * 32 + lane));
}

__device__ __forceinline__ v8f wmma16(v16h a, v16h b, v8f c) {
  return __builtin_amdgcn_wmma_f32_16x16x32_f16(
      /*neg_a=*/false, a, /*neg_b=*/false, b,
      /*c_mod=*/(short)0, c, /*reuse_a=*/false, /*reuse_b=*/false);
}

// ---- gfx1250 async global->LDS staging (ASYNCcnt path) --------------------
#if __has_builtin(__builtin_amdgcn_global_load_async_to_lds_b128)
#define HAS_ASYNC_LDS 1
#else
#define HAS_ASYNC_LDS 0
#endif

__device__ __forceinline__ void async_cp16(_Float16* l, const _Float16* g) {
#if HAS_ASYNC_LDS
  // Builtin signature (from hipcc diagnostic): param0 = global int4*, then
  // lds pointer, immediate offset, immediate cpol.
  __builtin_amdgcn_global_load_async_to_lds_b128(
      (__attribute__((address_space(1))) v4i*)g,
      (__attribute__((address_space(3))) v4i*)l, 0, 0);
#else
  *reinterpret_cast<v8h*>(l) = *reinterpret_cast<const v8h*>(g);
#endif
}

__device__ __forceinline__ void wait_async_then_barrier() {
#if HAS_ASYNC_LDS
#if __has_builtin(__builtin_amdgcn_s_wait_asynccnt)
  __builtin_amdgcn_s_wait_asynccnt(0);
#else
  asm volatile("s_wait_asynccnt 0x0" ::: "memory");
#endif
#endif
  __syncthreads();
}

// Cooperatively stage `halves` _Float16 (multiple of 8*blockDim) into LDS.
__device__ __forceinline__ void stage_weights(_Float16* dst, const _Float16* src,
                                              int halves) {
  for (int i = threadIdx.x; i < halves / 8; i += blockDim.x)
    async_cp16(dst + i * 8, src + i * 8);
}

// ---------------------------------------------------------------------------
// Weight packing: W is (din x 128) row-major f32 -> per-lane B fragments f16.
// B layout (16-bit 32x16): lanes 0-15 hold K=kc*32+{0..7,16..23} at N=lane,
// lanes 16-31 hold K=kc*32+{8..15,24..31} at N=lane-16.
// ---------------------------------------------------------------------------
__global__ void pack_weight_kernel(const float* __restrict__ W, int din, int kcN,
                                   _Float16* __restrict__ out) {
  int idx = blockIdx.x * blockDim.x + threadIdx.x;
  int total = kcN * NT * FRAG_H;
  if (idx >= total) return;
  int i    = idx & 15;
  int lane = (idx >> 4) & 31;
  int nt   = (idx >> 9) & 7;
  int kc   = idx >> 12;
  int k = kc * 32 + ((lane >= 16) ? 8 : 0) + ((i < 8) ? i : 16 + (i - 8));
  int n = nt * 16 + (lane & 15);
  out[idx] = (k < din) ? (_Float16)W[k * HID + n] : (_Float16)0.0f;
}

// h = x[:,0:2] @ ein_w + ein_b ; coord = x[:,2:5]
__global__ void embed_kernel(const float* __restrict__ x,
                             const float* __restrict__ ein_w,
                             const float* __restrict__ ein_b,
                             float* __restrict__ h, _Float16* __restrict__ h16,
                             float* __restrict__ coord) {
  int idx = blockIdx.x * blockDim.x + threadIdx.x;
  if (idx >= N_NODES * HID) return;
  int n = idx / HID, f = idx % HID;
  float v = x[n * 5 + 0] * ein_w[f] + x[n * 5 + 1] * ein_w[HID + f] + ein_b[f];
  h[idx] = v;
  h16[idx] = (_Float16)v;
  if (f < 3) coord[n * 3 + f] = x[n * 5 + 2 + f];
}

__global__ void deg_kernel(const int* __restrict__ ei, float* __restrict__ deg) {
  int e = blockIdx.x * blockDim.x + threadIdx.x;
  if (e < N_EDGES) atomicAdd(&deg[ei[e]], 1.0f);
}

__global__ void cnt_kernel(const int* __restrict__ batch, float* __restrict__ cnt) {
  int n = blockIdx.x * blockDim.x + threadIdx.x;
  if (n < N_NODES) atomicAdd(&cnt[batch[n]], 1.0f);
}

// ---------------------------------------------------------------------------
// Edge kernel: one wave == 16 edges; 4 waves/block. We1 pack staged into LDS
// asynchronously (72 KB, reused by every wave/tile of the block).
//   m1 = silu(e_in @ We1 + be1)   (K = 288 padded, 9 kchunks)
//   m  = silu(m1  @ We2 + be2)    -> agg atomics
//   mc = silu(m   @ Wc1 + bc1)
//   p  = mc @ Wc2 (per-edge scalar) -> trans atomics with cd
// ---------------------------------------------------------------------------
__global__ __launch_bounds__(WPB * 32) void edge_kernel(
    const int* __restrict__ ei, const float* __restrict__ ea,
    const float* __restrict__ coord, const _Float16* __restrict__ h16,
    const _Float16* __restrict__ We1p, const float* __restrict__ be1,
    const _Float16* __restrict__ We2p, const float* __restrict__ be2,
    const _Float16* __restrict__ Wc1p, const float* __restrict__ bc1,
    const float* __restrict__ Wc2,
    float* __restrict__ agg, float* __restrict__ trans_sum) {
  __shared__ __align__(32) _Float16 sWe1[9 * NT * FRAG_H];  // 73728 B
  __shared__ __align__(16) _Float16 lds[WPB][16][HID];      // 16384 B

  stage_weights(sWe1, We1p, 9 * NT * FRAG_H);

  const int wave = threadIdx.x >> 5;
  const int lane = threadIdx.x & 31;
  const int mrow = lane & 15;
  const bool hi  = lane >= 16;
  const int off0 = hi ? 8 : 0;
  const int ebase = (blockIdx.x * WPB + wave) * 16;
  const int e = ebase + mrow;
  const int r = ei[e];
  const int c = ei[N_EDGES + e];
  const float cdx = coord[3 * r + 0] - coord[3 * c + 0];
  const float cdy = coord[3 * r + 1] - coord[3 * c + 1];
  const float cdz = coord[3 * r + 2] - coord[3 * c + 2];
  const float radial = cdx * cdx + cdy * cdy + cdz * cdz;

  __builtin_prefetch(We2p, 0, 1);
  __builtin_prefetch(Wc1p, 0, 1);
  wait_async_then_barrier();   // We1 fragments now resident in LDS

  // ---- GEMM1: e_in @ We1 (B from LDS) ----
  v8f acc[NT] = {};
#pragma unroll
  for (int kc = 0; kc < 9; ++kc) {
    Frag a;
    if (kc < 4) {
      const _Float16* p = h16 + r * HID + kc * 32 + off0;
      a.h[0] = ld8h(p); a.h[1] = ld8h(p + 16);
    } else if (kc < 8) {
      const _Float16* p = h16 + c * HID + (kc - 4) * 32 + off0;
      a.h[0] = ld8h(p); a.h[1] = ld8h(p + 16);
    } else {
#pragma unroll
      for (int i = 0; i < 16; ++i) a.e[i] = (_Float16)0.0f;
      if (!hi) {
        a.e[0] = (_Float16)radial;
#pragma unroll
        for (int j = 0; j < N_EF; ++j) a.e[1 + j] = (_Float16)ea[e * N_EF + j];
      }
    }
#pragma unroll
    for (int nt = 0; nt < NT; ++nt)
      acc[nt] = wmma16(a.v, ldB(sWe1, kc, nt, lane), acc[nt]);
  }
  // epilogue: bias + silu -> LDS (f16, row-major m1 tile)
#pragma unroll
  for (int nt = 0; nt < NT; ++nt) {
    const float b = be1[nt * 16 + mrow];
#pragma unroll
    for (int v = 0; v < 8; ++v) {
      const int M = v + (hi ? 8 : 0);
      lds[wave][M][nt * 16 + mrow] = (_Float16)silu_f(acc[nt][v] + b);
    }
  }
  __syncthreads();

  Frag ma[4];
#pragma unroll
  for (int kc = 0; kc < 4; ++kc) {
    const _Float16* p = &lds[wave][mrow][kc * 32 + off0];
    ma[kc].h[0] = ld8h(p); ma[kc].h[1] = ld8h(p + 16);
  }
  __syncthreads();

  // ---- GEMM2: m = silu(m1 @ We2 + be2) ----
  v8f acc2[NT] = {};
#pragma unroll
  for (int kc = 0; kc < 4; ++kc)
#pragma unroll
    for (int nt = 0; nt < NT; ++nt)
      acc2[nt] = wmma16(ma[kc].v, ldB(We2p, kc, nt, lane), acc2[nt]);

  float mf[NT][8];
#pragma unroll
  for (int nt = 0; nt < NT; ++nt) {
    const float b = be2[nt * 16 + mrow];
#pragma unroll
    for (int v = 0; v < 8; ++v) mf[nt][v] = silu_f(acc2[nt][v] + b);
  }
  // agg scatter-add + stage m in LDS for next A relayout
#pragma unroll
  for (int v = 0; v < 8; ++v) {
    const int M = v + (hi ? 8 : 0);
    const int node = ei[ebase + M];
#pragma unroll
    for (int nt = 0; nt < NT; ++nt) {
      atomicAdd(&agg[node * HID + nt * 16 + mrow], mf[nt][v]);
      lds[wave][M][nt * 16 + mrow] = (_Float16)mf[nt][v];
    }
  }
  __syncthreads();
#pragma unroll
  for (int kc = 0; kc < 4; ++kc) {
    const _Float16* p = &lds[wave][mrow][kc * 32 + off0];
    ma[kc].h[0] = ld8h(p); ma[kc].h[1] = ld8h(p + 16);
  }
  __syncthreads();

  // ---- GEMM3: mc = silu(m @ Wc1 + bc1) ----
  v8f acc3[NT] = {};
#pragma unroll
  for (int kc = 0; kc < 4; ++kc)
#pragma unroll
    for (int nt = 0; nt < NT; ++nt)
      acc3[nt] = wmma16(ma[kc].v, ldB(Wc1p, kc, nt, lane), acc3[nt]);
#pragma unroll
  for (int nt = 0; nt < NT; ++nt) {
    const float b = bc1[nt * 16 + mrow];
#pragma unroll
    for (int v = 0; v < 8; ++v) {
      const int M = v + (hi ? 8 : 0);
      lds[wave][M][nt * 16 + mrow] = (_Float16)silu_f(acc3[nt][v] + b);
    }
  }
  __syncthreads();

  // ---- p = mc_row . Wc2 ; trans scatter-add (lanes 0-15 own one edge each)
  if (!hi) {
    float p = 0.0f;
#pragma unroll 16
    for (int k = 0; k < HID; ++k) p += (float)lds[wave][mrow][k] * Wc2[k];
    atomicAdd(&trans_sum[3 * r + 0], cdx * p);
    atomicAdd(&trans_sum[3 * r + 1], cdy * p);
    atomicAdd(&trans_sum[3 * r + 2], cdz * p);
  }
}

// ---------------------------------------------------------------------------
// Node kernel: one wave == 16 nodes; 4 waves/block. Wn1 pack staged into LDS.
//   out = silu([h | agg] @ Wn1 + bn1) @ Wn2 + bn2 ; h' = h + out
// ---------------------------------------------------------------------------
__global__ __launch_bounds__(WPB * 32) void node_kernel(
    const float* __restrict__ h, const _Float16* __restrict__ h16,
    const float* __restrict__ agg,
    const _Float16* __restrict__ Wn1p, const float* __restrict__ bn1,
    const _Float16* __restrict__ Wn2p, const float* __restrict__ bn2,
    float* __restrict__ h_out, _Float16* __restrict__ h16_out) {
  __shared__ __align__(32) _Float16 sWn1[8 * NT * FRAG_H];  // 65536 B
  __shared__ __align__(16) _Float16 lds[WPB][16][HID];

  stage_weights(sWn1, Wn1p, 8 * NT * FRAG_H);

  const int wave = threadIdx.x >> 5;
  const int lane = threadIdx.x & 31;
  const int mrow = lane & 15;
  const bool hi  = lane >= 16;
  const int off0 = hi ? 8 : 0;
  const int nbase = (blockIdx.x * WPB + wave) * 16;
  const int node = nbase + mrow;

  __builtin_prefetch(Wn2p, 0, 1);
  wait_async_then_barrier();

  v8f acc[NT] = {};
#pragma unroll
  for (int kc = 0; kc < 8; ++kc) {
    Frag a;
    if (kc < 4) {
      const _Float16* p = h16 + node * HID + kc * 32 + off0;
      a.h[0] = ld8h(p); a.h[1] = ld8h(p + 16);
    } else {
      const float* q = agg + node * HID + (kc - 4) * 32 + off0;
#pragma unroll
      for (int i = 0; i < 8; ++i) {
        a.e[i]     = (_Float16)q[i];
        a.e[8 + i] = (_Float16)q[16 + i];
      }
    }
#pragma unroll
    for (int nt = 0; nt < NT; ++nt)
      acc[nt] = wmma16(a.v, ldB(sWn1, kc, nt, lane), acc[nt]);
  }
#pragma unroll
  for (int nt = 0; nt < NT; ++nt) {
    const float b = bn1[nt * 16 + mrow];
#pragma unroll
    for (int v = 0; v < 8; ++v) {
      const int M = v + (hi ? 8 : 0);
      lds[wave][M][nt * 16 + mrow] = (_Float16)silu_f(acc[nt][v] + b);
    }
  }
  __syncthreads();
  Frag ma[4];
#pragma unroll
  for (int kc = 0; kc < 4; ++kc) {
    const _Float16* p = &lds[wave][mrow][kc * 32 + off0];
    ma[kc].h[0] = ld8h(p); ma[kc].h[1] = ld8h(p + 16);
  }
  __syncthreads();

  v8f acc2[NT] = {};
#pragma unroll
  for (int kc = 0; kc < 4; ++kc)
#pragma unroll
    for (int nt = 0; nt < NT; ++nt)
      acc2[nt] = wmma16(ma[kc].v, ldB(Wn2p, kc, nt, lane), acc2[nt]);

#pragma unroll
  for (int v = 0; v < 8; ++v) {
    const int n2 = nbase + v + (hi ? 8 : 0);
#pragma unroll
    for (int nt = 0; nt < NT; ++nt) {
      const int f = nt * 16 + mrow;
      const float val = h[n2 * HID + f] + acc2[nt][v] + bn2[f];
      h_out[n2 * HID + f] = val;
      h16_out[n2 * HID + f] = (_Float16)val;
    }
  }
}

// out = h16 @ W + bias (K = 128, no activation) — eout projection.
__global__ __launch_bounds__(256) void gemm_out_kernel(
    const _Float16* __restrict__ h16, const _Float16* __restrict__ Wp,
    const float* __restrict__ bias, float* __restrict__ out) {
  const int wave = threadIdx.x >> 5;
  const int lane = threadIdx.x & 31;
  const int mrow = lane & 15;
  const bool hi  = lane >= 16;
  const int off0 = hi ? 8 : 0;
  const int nbase = (blockIdx.x * 8 + wave) * 16;
  const int node = nbase + mrow;

  v8f acc[NT] = {};
#pragma unroll
  for (int kc = 0; kc < 4; ++kc) {
    Frag a;
    const _Float16* p = h16 + node * HID + kc * 32 + off0;
    a.h[0] = ld8h(p); a.h[1] = ld8h(p + 16);
#pragma unroll
    for (int nt = 0; nt < NT; ++nt)
      acc[nt] = wmma16(a.v, ldB(Wp, kc, nt, lane), acc[nt]);
  }
#pragma unroll
  for (int v = 0; v < 8; ++v) {
    const int n2 = nbase + v + (hi ? 8 : 0);
#pragma unroll
    for (int nt = 0; nt < NT; ++nt) {
      const int f = nt * 16 + mrow;
      out[n2 * HID + f] = acc[nt][v] + bias[f];
    }
  }
}

__global__ void coord_update_kernel(float* __restrict__ coord,
                                    const float* __restrict__ trans_sum,
                                    const float* __restrict__ deg) {
  int idx = blockIdx.x * blockDim.x + threadIdx.x;
  if (idx >= N_NODES * 3) return;
  int n = idx / 3;
  coord[idx] += trans_sum[idx] / fmaxf(deg[n], 1.0f);
}

__global__ void pool_h_kernel(const float* __restrict__ hout,
                              const int* __restrict__ batch,
                              float* __restrict__ pooled) {
  int idx = blockIdx.x * blockDim.x + threadIdx.x;
  if (idx >= N_NODES * HID) return;
  int n = idx / HID, f = idx % HID;
  atomicAdd(&pooled[batch[n] * (HID + 3) + 3 + f], hout[idx]);
}

__global__ void pool_c_kernel(const float* __restrict__ coord,
                              const int* __restrict__ batch,
                              float* __restrict__ pooled) {
  int idx = blockIdx.x * blockDim.x + threadIdx.x;
  if (idx >= N_NODES * 3) return;
  int n = idx / 3, d = idx % 3;
  atomicAdd(&pooled[batch[n] * (HID + 3) + d], coord[idx]);
}

__global__ void head_kernel(const float* __restrict__ pooled,
                            const float* __restrict__ cnt,
                            const float* __restrict__ lin_w,
                            const float* __restrict__ lin_b,
                            float* __restrict__ out) {
  int idx = blockIdx.x * blockDim.x + threadIdx.x;
  if (idx >= N_GRAPHS * N_PRED) return;
  int g = idx / N_PRED, p = idx % N_PRED;
  float inv = 1.0f / fmaxf(cnt[g], 1.0f);
  float s = lin_b[p];
  for (int f = 0; f < HID + 3; ++f)
    s += pooled[g * (HID + 3) + f] * inv * lin_w[f * N_PRED + p];
  out[idx] = s;
}

// ---------------------------------------------------------------------------
extern "C" void kernel_launch(void* const* d_in, const int* in_sizes, int n_in,
                              void* d_out, int out_size, void* d_ws, size_t ws_size,
                              hipStream_t stream) {
  (void)in_sizes; (void)n_in; (void)out_size; (void)ws_size;

  const float* x       = (const float*)d_in[0];
  const int*   ei      = (const int*)d_in[1];
  const float* ea      = (const float*)d_in[2];
  const int*   batch   = (const int*)d_in[3];
  const float* ein_b   = (const float*)d_in[4];
  const float* ein_w   = (const float*)d_in[5];
  const float* eout_b  = (const float*)d_in[6];
  const float* eout_w  = (const float*)d_in[7];
  const float *Wc1[N_LAYERS], *Wc2[N_LAYERS], *We1[N_LAYERS], *We2[N_LAYERS];
  const float *Wn1[N_LAYERS], *Wn2[N_LAYERS];
  const float *bc1[N_LAYERS], *be1[N_LAYERS], *be2[N_LAYERS], *bn1[N_LAYERS], *bn2[N_LAYERS];
  for (int i = 0; i < N_LAYERS; ++i) {
    int b = 8 + i * 11;
    Wc1[i] = (const float*)d_in[b + 0];
    Wc2[i] = (const float*)d_in[b + 1];
    We1[i] = (const float*)d_in[b + 2];
    We2[i] = (const float*)d_in[b + 3];
    Wn1[i] = (const float*)d_in[b + 4];
    Wn2[i] = (const float*)d_in[b + 5];
    bc1[i] = (const float*)d_in[b + 6];
    be1[i] = (const float*)d_in[b + 7];
    be2[i] = (const float*)d_in[b + 8];
    bn1[i] = (const float*)d_in[b + 9];
    bn2[i] = (const float*)d_in[b + 10];
  }
  const float* lin_b = (const float*)d_in[52];
  const float* lin_w = (const float*)d_in[53];
  float* out = (float*)d_out;

  // ---- carve workspace (~22 MB) ----
  char* base = (char*)d_ws;
  size_t off = 0;
  auto carve = [&](size_t bytes) -> char* {
    off = (off + 255) & ~(size_t)255;
    char* p = base + off;
    off += bytes;
    return p;
  };
  float*     hA     = (float*)carve((size_t)N_NODES * HID * 4);
  float*     hB     = (float*)carve((size_t)N_NODES * HID * 4);
  _Float16*  h16A   = (_Float16*)carve((size_t)N_NODES * HID * 2);
  _Float16*  h16B   = (_Float16*)carve((size_t)N_NODES * HID * 2);
  float*     agg    = (float*)carve((size_t)N_NODES * HID * 4);
  float*     coord  = (float*)carve((size_t)N_NODES * 3 * 4);
  float*     trans  = (float*)carve((size_t)N_NODES * 3 * 4);
  float*     deg    = (float*)carve((size_t)N_NODES * 4);
  float*     cnt    = (float*)carve((size_t)N_GRAPHS * 4);
  float*     hout   = (float*)carve((size_t)N_NODES * HID * 4);
  float*     pooled = (float*)carve((size_t)N_GRAPHS * (HID + 3) * 4);
  _Float16 *We1p[N_LAYERS], *We2p[N_LAYERS], *Wn1p[N_LAYERS], *Wn2p[N_LAYERS], *Wc1p[N_LAYERS];
  const size_t fragBytes = (size_t)NT * FRAG_H * 2;  // per kchunk
  for (int i = 0; i < N_LAYERS; ++i) {
    We1p[i] = (_Float16*)carve(9 * fragBytes);
    We2p[i] = (_Float16*)carve(4 * fragBytes);
    Wn1p[i] = (_Float16*)carve(8 * fragBytes);
    Wn2p[i] = (_Float16*)carve(4 * fragBytes);
    Wc1p[i] = (_Float16*)carve(4 * fragBytes);
  }
  _Float16* eoutp = (_Float16*)carve(4 * fragBytes);

  // ---- prep: pack weights, embed, counts ----
  auto pack = [&](const float* W, int din, int kcN, _Float16* dst) {
    int total = kcN * NT * FRAG_H;
    pack_weight_kernel<<<(total + 255) / 256, 256, 0, stream>>>(W, din, kcN, dst);
  };
  for (int i = 0; i < N_LAYERS; ++i) {
    pack(We1[i], 2 * HID + 1 + N_EF, 9, We1p[i]);
    pack(We2[i], HID, 4, We2p[i]);
    pack(Wn1[i], 2 * HID, 8, Wn1p[i]);
    pack(Wn2[i], HID, 4, Wn2p[i]);
    pack(Wc1[i], HID, 4, Wc1p[i]);
  }
  pack(eout_w, HID, 4, eoutp);

  embed_kernel<<<(N_NODES * HID + 255) / 256, 256, 0, stream>>>(x, ein_w, ein_b, hA, h16A, coord);
  (void)hipMemsetAsync(deg, 0, (size_t)N_NODES * 4, stream);
  (void)hipMemsetAsync(cnt, 0, (size_t)N_GRAPHS * 4, stream);
  deg_kernel<<<(N_EDGES + 255) / 256, 256, 0, stream>>>(ei, deg);
  cnt_kernel<<<(N_NODES + 255) / 256, 256, 0, stream>>>(batch, cnt);

  // ---- layers ----
  float* h_cur = hA;        float* h_nxt = hB;
  _Float16* h16_cur = h16A; _Float16* h16_nxt = h16B;
  for (int i = 0; i < N_LAYERS; ++i) {
    (void)hipMemsetAsync(agg, 0, (size_t)N_NODES * HID * 4, stream);
    (void)hipMemsetAsync(trans, 0, (size_t)N_NODES * 3 * 4, stream);
    edge_kernel<<<N_EDGES / (16 * WPB), WPB * 32, 0, stream>>>(
        ei, ea, coord, h16_cur,
        We1p[i], be1[i], We2p[i], be2[i], Wc1p[i], bc1[i], Wc2[i],
        agg, trans);
    coord_update_kernel<<<(N_NODES * 3 + 255) / 256, 256, 0, stream>>>(coord, trans, deg);
    node_kernel<<<N_NODES / (16 * WPB), WPB * 32, 0, stream>>>(
        h_cur, h16_cur, agg, Wn1p[i], bn1[i], Wn2p[i], bn2[i], h_nxt, h16_nxt);
    float* tf = h_cur; h_cur = h_nxt; h_nxt = tf;
    _Float16* th = h16_cur; h16_cur = h16_nxt; h16_nxt = th;
  }

  // ---- readout ----
  gemm_out_kernel<<<N_NODES / 128, 256, 0, stream>>>(h16_cur, eoutp, eout_b, hout);
  (void)hipMemsetAsync(pooled, 0, (size_t)N_GRAPHS * (HID + 3) * 4, stream);
  pool_h_kernel<<<(N_NODES * HID + 255) / 256, 256, 0, stream>>>(hout, batch, pooled);
  pool_c_kernel<<<(N_NODES * 3 + 255) / 256, 256, 0, stream>>>(coord, batch, pooled);
  head_kernel<<<(N_GRAPHS * N_PRED + 255) / 256, 256, 0, stream>>>(pooled, cnt, lin_w, lin_b, out);
}
